// FastRopeAttention_62869731279200
// MI455X (gfx1250) — matrix operand order repaired and checked by reading, not compiled, not run on hardware
//
#include <hip/hip_runtime.h>
#include <stdint.h>

// ---------------------------------------------------------------------------
// FastRopeAttention for MI455X (gfx1250), wave32 + v_wmma_f32_16x16x32_bf16.
// Pipeline: [GEMM Q][GEMM KV] -> [RoPE+scale] -> [flash attention] -> [GEMM O]
// Flash attention computes score/output tiles TRANSPOSED so each lane owns a
// single query: softmax stats are in-lane + one shfl_xor(16). K-tile staging
// uses GLOBAL_LOAD_ASYNC_TO_LDS_B128 (ASYNCcnt) when the toolchain has it.
// ---------------------------------------------------------------------------

namespace {
constexpr int B = 4, S = 2048, D = 1024, H = 16, P = 64;
constexpr int HP  = H * P;       // 1024
constexpr int KVW = 2 * HP;      // 2048
constexpr int M   = B * S;       // 8192
}

typedef __attribute__((ext_vector_type(16))) __bf16          v16bf;
typedef __attribute__((ext_vector_type(8)))  float           v8f;
typedef __attribute__((ext_vector_type(8)))  unsigned short  ushort8;

#if defined(__has_builtin)
#  if __has_builtin(__builtin_amdgcn_global_load_async_to_lds_b128) && \
      __has_builtin(__builtin_amdgcn_s_wait_asynccnt)
#    define USE_ASYNC_LDS 1
#  endif
#endif
#ifndef USE_ASYNC_LDS
#  define USE_ASYNC_LDS 0
#endif

union V16 { v16bf v; ushort8 h[2]; unsigned short s[16]; };

__device__ __forceinline__ unsigned short f2bf(float f) {
  union { float f; unsigned u; } x; x.f = f;
  unsigned r = x.u + 0x7FFFu + ((x.u >> 16) & 1u);   // round-nearest-even
  return (unsigned short)(r >> 16);
}
__device__ __forceinline__ float bf2f(unsigned short h) {
  union { unsigned u; float f; } x; x.u = ((unsigned)h) << 16; return x.f;
}

// A-fragment (16-bit, 16x32): lane holds row M=lane&15; element e<8 -> K=c0+e,
// e>=8 -> K=c0+16+(e-8), where c0 = kBase + (lane>>4)*8. Two 16B loads.
__device__ __forceinline__ v16bf ldfragA(const unsigned short* rowBase, int c0) {
  V16 u;
  u.h[0] = *(const ushort8*)(rowBase + c0);
  u.h[1] = *(const ushort8*)(rowBase + c0 + 16);
  return u.v;
}
// B-fragment (16-bit, 32x16): lane holds column N=lane&15; element e<8 -> K=c0+e,
// e>=8 -> K=c0+16+(e-8) (same K order as A), rows strided by 'stride'.
__device__ __forceinline__ v16bf ldfragB(const unsigned short* p, int c0, int stride) {
  V16 u;
  #pragma unroll
  for (int e = 0; e < 8; ++e) {
    u.s[e]     = p[(c0 + e) * stride];
    u.s[e + 8] = p[(c0 + 16 + e) * stride];
  }
  return u.v;
}
__device__ __forceinline__ v8f wmma_bf16(v16bf a, v16bf b, v8f c) {
  return __builtin_amdgcn_wmma_f32_16x16x32_bf16(false, a, false, b, (short)0, c,
                                                 false, false);
}

// ---------------------------------------------------------------------------
// Tiled GEMM: out[M,N] = A[M,K] (f32 or bf16) * W[K,N] (f32), out f32 or bf16.
// 256 threads = 8 waves; 128x64 tile per workgroup; K stepped by 32 via LDS.
// ---------------------------------------------------------------------------
template<bool A_BF16, bool OUT_F32>
__global__ __launch_bounds__(256)
void gemm_wmma_kernel(const void* __restrict__ A_, const float* __restrict__ W,
                      void* __restrict__ out_, int Mdim, int Ndim, int Kdim)
{
  __shared__ unsigned short Xs[128 * 40];  // [row][32 k + pad], 16B-aligned runs
  __shared__ unsigned short Ws[32 * 72];   // [k][64 n + pad]
  const int m0 = blockIdx.x * 128;
  const int n0 = blockIdx.y * 64;
  const int t  = threadIdx.x;
  const int lane = t & 31;
  const int w    = t >> 5;
  const int li   = lane & 15;
  const int hi   = lane >> 4;

  v8f acc[4] = {};

  for (int k0 = 0; k0 < Kdim; k0 += 32) {
    {  // stage A tile 128x32 (convert f32->bf16 on the fly)
      const int row = t >> 1;
      const int cb  = (t & 1) * 16;
      unsigned short* dst = &Xs[row * 40 + cb];
      if constexpr (A_BF16) {
        const unsigned short* g =
            (const unsigned short*)A_ + (size_t)(m0 + row) * Kdim + k0 + cb;
        *(ushort8*)(dst)     = *(const ushort8*)(g);
        *(ushort8*)(dst + 8) = *(const ushort8*)(g + 8);
      } else {
        const float* g = (const float*)A_ + (size_t)(m0 + row) * Kdim + k0 + cb;
        #pragma unroll
        for (int i = 0; i < 16; ++i) dst[i] = f2bf(g[i]);
      }
    }
    {  // stage W tile 32x64
      const int kk  = t >> 3;
      const int nb8 = (t & 7) * 8;
      const float* g = W + (size_t)(k0 + kk) * Ndim + n0 + nb8;
      unsigned short* dst = &Ws[kk * 72 + nb8];
      #pragma unroll
      for (int i = 0; i < 8; ++i) dst[i] = f2bf(g[i]);
    }
    __syncthreads();

    const v16bf a = ldfragA(&Xs[(w * 16 + li) * 40], hi * 8);
    #pragma unroll
    for (int nb = 0; nb < 4; ++nb) {
      const v16bf bfr = ldfragB(&Ws[nb * 16 + li], hi * 8, 72);
      acc[nb] = wmma_bf16(a, bfr, acc[nb]);
    }
    __syncthreads();
  }

  #pragma unroll
  for (int nb = 0; nb < 4; ++nb)
    #pragma unroll
    for (int r = 0; r < 8; ++r) {
      const int row = m0 + w * 16 + r + 8 * hi;   // C/D layout: M = r + 8*(lane>>4)
      const int col = n0 + nb * 16 + li;          //             N = lane&15
      const float v = acc[nb][r];
      if constexpr (OUT_F32)
        ((float*)out_)[(size_t)row * Ndim + col] = v;
      else
        ((unsigned short*)out_)[(size_t)row * Ndim + col] = f2bf(v);
    }
}

// ---------------------------------------------------------------------------
// Partial RoPE (first 32 of 64 head channels), in-place on bf16 Q / K.
// Q additionally scaled by 1/sqrt(P)=0.125 (folds attention scale into Q).
// ---------------------------------------------------------------------------
__global__ __launch_bounds__(256)
void rope_scale_kernel(unsigned short* __restrict__ Qbf,
                       unsigned short* __restrict__ KVbf,
                       const int* __restrict__ posOffPtr)
{
  const int idx = blockIdx.x * blockDim.x + threadIdx.x;
  if (idx >= B * S * H) return;
  const int h = idx % H;
  const int s = (idx / H) % S;
  const int b = idx / (H * S);
  const float c = 0.57564627324851142f;  // ln(10000)/16

  {  // Q: rope + scale
    unsigned short* q = Qbf + (size_t)(b * S + s) * HP + h * P;
    const float pos = (float)(posOffPtr[0] + s);
    #pragma unroll
    for (int j = 0; j < 16; ++j) {
      const float ang = pos * __expf(-c * (float)j);
      const float sn = __sinf(ang), cs = __cosf(ang);
      const float x0 = bf2f(q[j]), x1 = bf2f(q[j + 16]);
      q[j]      = f2bf((x0 * cs - x1 * sn) * 0.125f);
      q[j + 16] = f2bf((x1 * cs + x0 * sn) * 0.125f);
    }
    #pragma unroll
    for (int j = 32; j < 64; ++j) q[j] = f2bf(bf2f(q[j]) * 0.125f);
  }
  {  // K: rope only
    unsigned short* k = KVbf + (size_t)(b * S + s) * KVW + h * P;
    const float pos = (float)s;
    #pragma unroll
    for (int j = 0; j < 16; ++j) {
      const float ang = pos * __expf(-c * (float)j);
      const float sn = __sinf(ang), cs = __cosf(ang);
      const float x0 = bf2f(k[j]), x1 = bf2f(k[j + 16]);
      k[j]      = f2bf(x0 * cs - x1 * sn);
      k[j + 16] = f2bf(x1 * cs + x0 * sn);
    }
  }
}

// ---------------------------------------------------------------------------
// Causal flash attention, transposed-tile formulation.
// Grid: B*H*(S/64) workgroups of 128 threads (4 waves); each wave owns a
// 16-query tile. Per 64-key block:
//   S^T = K * Q^T   (A = K block rows, natural LDS; B = Q^T, built once)
//   O^T += V^T * P  (A = V^T, scatter-staged; B = P key-major from D layout)
// Lane owns ONE query (N = lane&15): softmax stats are in-lane reductions
// plus a single shfl_xor(16); m/l/scale are per-lane scalars.
// K tile staged with async global->LDS (ASYNCcnt) when available.
// ---------------------------------------------------------------------------
__global__ __launch_bounds__(128)
void flash_attn_kernel(const unsigned short* __restrict__ Qbf,
                       const unsigned short* __restrict__ KVbf,
                       unsigned short* __restrict__ Obf)
{
  __shared__ unsigned short Qt[64 * 72];      // Q^T: [chan][query]   (once)
  __shared__ unsigned short Ks[64 * 72];      // K:   [key][chan]     (natural)
  __shared__ unsigned short Vt[64 * 72];      // V^T: [chan][key]
  __shared__ unsigned short Pl[4][64 * 24];   // per-wave P: [key][query]

  constexpr int NQ = S / 64;
  const int qblk = blockIdx.x % NQ;
  const int h    = (blockIdx.x / NQ) % H;
  const int b    = blockIdx.x / (NQ * H);
  const int t = threadIdx.x, lane = t & 31, w = t >> 5;
  const int li = lane & 15, hi = lane >> 4;
  const int qb0 = qblk * 64;

  {  // build Q^T tile once (RoPE'd + pre-scaled bf16)
    const int qy = t & 63;
    const int c0 = (t >> 6) * 32;
    const unsigned short* gq =
        Qbf + (size_t)(b * S + qb0 + qy) * HP + h * P + c0;
    #pragma unroll
    for (int cc = 0; cc < 32; ++cc) Qt[(c0 + cc) * 72 + qy] = gq[cc];
  }
  __syncthreads();

  float mrow = -1e30f, lrow = 0.0f;   // per-lane scalars (one query per lane)
  v8f acc[4] = {};                    // O^T: chans (rows) x this lane's query

  for (int kb = 0; kb <= qb0; kb += 64) {     // causal: stop at diagonal block
    {  // stage K (async direct-to-LDS if available) and V^T (scatter)
      const int key = t & 63;
      const int c0  = (t >> 6) * 32;
      const unsigned short* gk =
          KVbf + (size_t)(b * S + kb + key) * KVW + h * P + c0;
#if USE_ASYNC_LDS
      typedef __attribute__((vector_size(16))) int v4i_t;
      typedef __attribute__((address_space(1))) v4i_t gv4i_t;
      typedef __attribute__((address_space(3))) v4i_t lv4i_t;
      gv4i_t* gsrc = (gv4i_t*)gk;                     // global src (AS1)
      lv4i_t* ldst = (lv4i_t*)&Ks[key * 72 + c0];     // LDS dst (AS3)
      #pragma unroll
      for (int i = 0; i < 4; ++i)
        __builtin_amdgcn_global_load_async_to_lds_b128(gsrc + i, ldst + i, 0, 0);
#else
      ushort8* dk = (ushort8*)(&Ks[key * 72 + c0]);
      #pragma unroll
      for (int i = 0; i < 4; ++i) dk[i] = ((const ushort8*)gk)[i];
#endif
      const unsigned short* gv = gk + HP;
      #pragma unroll
      for (int cc = 0; cc < 32; ++cc) Vt[(c0 + cc) * 72 + key] = gv[cc];
      if (kb + 64 <= qb0) {  // prefetch next key block while this one computes
        __builtin_prefetch(gk + (size_t)64 * KVW, 0, 1);
        __builtin_prefetch(gv + (size_t)64 * KVW, 0, 1);
      }
    }
#if USE_ASYNC_LDS
    __builtin_amdgcn_s_wait_asynccnt(0);
#endif
    __syncthreads();

    // S^T = K * Q^T : 4 key sub-blocks (M) x this wave's 16 queries (N)
    float sc[4][8];
    #pragma unroll
    for (int mb = 0; mb < 4; ++mb) {
      v8f sv = {};
      sv = wmma_bf16(ldfragA(&Ks[(mb * 16 + li) * 72], 0 + hi * 8),
                     ldfragB(&Qt[w * 16 + li], hi * 8, 72), sv);
      sv = wmma_bf16(ldfragA(&Ks[(mb * 16 + li) * 72], 32 + hi * 8),
                     ldfragB(&Qt[32 * 72 + w * 16 + li], hi * 8, 72), sv);
      #pragma unroll
      for (int r = 0; r < 8; ++r) sc[mb][r] = sv[r];
    }
    if (kb == qb0) {  // diagonal block: mask key > query
      #pragma unroll
      for (int mb = 0; mb < 4; ++mb)
        #pragma unroll
        for (int r = 0; r < 8; ++r)
          if (mb * 16 + r + 8 * hi > w * 16 + li) sc[mb][r] = -1e30f;
    }

    // online softmax: in-lane reduce over 32 keys + one half-swap shuffle
    float vmax = sc[0][0];
    #pragma unroll
    for (int mb = 0; mb < 4; ++mb)
      #pragma unroll
      for (int r = 0; r < 8; ++r) vmax = fmaxf(vmax, sc[mb][r]);
    vmax = fmaxf(vmax, __shfl_xor(vmax, 16, 32));
    const float mn   = fmaxf(mrow, vmax);
    const float sOld = __expf(mrow - mn);
    mrow = mn;

    unsigned short* pw = &Pl[w][0];
    float rs = 0.0f;
    #pragma unroll
    for (int mb = 0; mb < 4; ++mb)
      #pragma unroll
      for (int r = 0; r < 8; ++r) {
        const float p = __expf(sc[mb][r] - mrow);
        rs += p;
        pw[(mb * 16 + r + 8 * hi) * 24 + li] = f2bf(p);  // P key-major
      }
    rs += __shfl_xor(rs, 16, 32);
    lrow = lrow * sOld + rs;
    #pragma unroll
    for (int nb = 0; nb < 4; ++nb) acc[nb] *= sOld;   // scalar splat rescale

    __syncthreads();  // Pl visible; waves stay in step

    // O^T += V^T * P : 4 chan sub-blocks x this lane's query
    #pragma unroll
    for (int nb = 0; nb < 4; ++nb) {
      acc[nb] = wmma_bf16(ldfragA(&Vt[(nb * 16 + li) * 72], 0 + hi * 8),
                          ldfragB(&pw[li], hi * 8, 24), acc[nb]);
      acc[nb] = wmma_bf16(ldfragA(&Vt[(nb * 16 + li) * 72], 32 + hi * 8),
                          ldfragB(&pw[32 * 24 + li], hi * 8, 24), acc[nb]);
    }
    __syncthreads();  // protect K/V tiles before next stage
  }

  // epilogue: lane owns query qb0+w*16+li; chans nb*16 + 8*hi + r contiguous
  const float linv = 1.0f / lrow;
  unsigned short* orow =
      Obf + (size_t)(b * S + qb0 + w * 16 + li) * HP + h * P;
  #pragma unroll
  for (int nb = 0; nb < 4; ++nb) {
    ushort8 pack;
    #pragma unroll
    for (int r = 0; r < 8; ++r) pack[r] = f2bf(acc[nb][r] * linv);
    *(ushort8*)(orow + nb * 16 + 8 * hi) = pack;
  }
}

// ---------------------------------------------------------------------------
extern "C" void kernel_launch(void* const* d_in, const int* in_sizes, int n_in,
                              void* d_out, int out_size, void* d_ws, size_t ws_size,
                              hipStream_t stream)
{
  (void)in_sizes; (void)n_in; (void)out_size; (void)ws_size;
  const float* curr = (const float*)d_in[0];  // curr_state [B,S,D]
  const float* att  = (const float*)d_in[1];  // attend_to  [B,S,D]
  // d_in[2]: causal mask (recomputed analytically on device) -- unused
  const float* Wq   = (const float*)d_in[3];  // [D, H*P]
  const float* Wkv  = (const float*)d_in[4];  // [D, 2*H*P]
  const float* Wo   = (const float*)d_in[5];  // [H*P, D]
  const int*   pos  = (const int*)d_in[6];    // pos_offset scalar

  unsigned short* Qbf  = (unsigned short*)d_ws;           // [M, HP]  bf16
  unsigned short* KVbf = Qbf  + (size_t)M * HP;           // [M, KVW] bf16
  unsigned short* Obf  = KVbf + (size_t)M * KVW;          // [M, HP]  bf16

  dim3 blk(256);
  gemm_wmma_kernel<false, false><<<dim3(M / 128, HP  / 64), blk, 0, stream>>>(curr, Wq,  Qbf,  M, HP,  D);
  gemm_wmma_kernel<false, false><<<dim3(M / 128, KVW / 64), blk, 0, stream>>>(att,  Wkv, KVbf, M, KVW, D);
  rope_scale_kernel<<<(B * S * H + 255) / 256, 256, 0, stream>>>(Qbf, KVbf, pos);
  flash_attn_kernel<<<B * H * (S / 64), 128, 0, stream>>>(Qbf, KVbf, Obf);
  gemm_wmma_kernel<true, true><<<dim3(M / 128, D / 64), blk, 0, stream>>>(Obf, Wo, d_out, M, D, HP);
}
